// SQVAE_17059610100523
// MI455X (gfx1250) — matrix operand order repaired and checked by reading, not compiled, not run on hardware
//
#include <hip/hip_runtime.h>

// ---------------------------------------------------------------------------
// SQVAE forward on gfx1250 (MI455X): split-bf16 WMMA for the big GEMMs,
// fused SNN recurrences, packed-bit spike storage.
// ---------------------------------------------------------------------------

typedef __attribute__((ext_vector_type(16))) __bf16 bf16x16;
typedef __attribute__((ext_vector_type(8)))  float  floatx8;

union FragU { bf16x16 v; uint4 q[2]; };

static constexpr int   T_STEPS = 64;
static constexpr float BETAv   = 0.95f;
static constexpr float THRv    = 1.0f;
static constexpr int   BATCH   = 1024;
static constexpr int   HID     = 320;     // hidden dim (K of layer-2 GEMMs)
static constexpr int   LAT     = 2048;    // latent dim
static constexpr int   NIN     = 40;      // input dim
static constexpr int   KCH     = HID / 32;  // 10 K-chunks of 32
static constexpr int   BM      = 64;        // M tile per block
static constexpr int   LDA     = 328;       // LDS row stride (bf16), 16B aligned, bank-spread

__device__ __forceinline__ unsigned short f2bf(float f) {
    unsigned u = __float_as_uint(f);
    unsigned r = (u + 0x7FFFu + ((u >> 16) & 1u)) >> 16;   // round-to-nearest-even
    return (unsigned short)r;
}
__device__ __forceinline__ float bf2f(unsigned short h) {
    return __uint_as_float(((unsigned)h) << 16);
}

__device__ __forceinline__ floatx8 wmma_bf16(bf16x16 a, bf16x16 b, floatx8 c) {
    // (neg_a, A, neg_b, B, c_mod, C, reuse_a, reuse_b)
    return __builtin_amdgcn_wmma_f32_16x16x32_bf16(false, a, false, b, (short)0, c,
                                                   false, false);
}

// A fragment (16x32 bf16, MxK): lanes 0-15 hold K {0..7,16..23}, lanes 16-31 K {8..15,24..31}
__device__ __forceinline__ bf16x16 loadA_lds(const unsigned short* lA, int mRow, int lane, int c) {
    const unsigned short* p = lA + (size_t)(mRow + (lane & 15)) * LDA
                                 + c * 32 + ((lane & 16) ? 8 : 0);
    FragU u;
    u.q[0] = *(const uint4*)p;
    u.q[1] = *(const uint4*)(p + 16);
    return u.v;
}
// B fragment (32x16 bf16, KxN): lane = N, lanes 0-15 K {0..7,16..23}, lanes 16-31 K {8..15,24..31}
__device__ __forceinline__ bf16x16 loadB_lds(const unsigned short* lW, int nRow, int lane, int c) {
    const unsigned short* p = lW + (size_t)(nRow + (lane & 15)) * LDA
                                 + c * 32 + ((lane & 16) ? 8 : 0);
    FragU u;
    u.q[0] = *(const uint4*)p;
    u.q[1] = *(const uint4*)(p + 16);
    return u.v;
}

// ---------------------------------------------------------------------------
// Prep: split f32 weights into bf16 hi + bf16 lo planes
// ---------------------------------------------------------------------------
__global__ void split_bf16_kernel(const float* __restrict__ src,
                                  unsigned short* __restrict__ hi,
                                  unsigned short* __restrict__ lo, int n) {
    int i = blockIdx.x * 256 + threadIdx.x;
    if (i >= n) return;
    float v = src[i];
    unsigned short h = f2bf(v);
    hi[i] = h;
    lo[i] = f2bf(v - bf2f(h));
}

// dec_W2 is [40][320]; pad to [48][320] with zero rows so N-tiles are clean
__global__ void split_pad48_kernel(const float* __restrict__ W,
                                   unsigned short* __restrict__ hi,
                                   unsigned short* __restrict__ lo) {
    int i = blockIdx.x * 256 + threadIdx.x;
    if (i >= 48 * HID) return;
    int r = i / HID;
    float v = (r < NIN) ? W[i] : 0.0f;
    unsigned short h = f2bf(v);
    hi[i] = h;
    lo[i] = f2bf(v - bf2f(h));
}

// ---------------------------------------------------------------------------
// Encoder fc1: cur1 = x @ W1^T + b1   (1024x40x320 — tiny, scalar f32)
// ---------------------------------------------------------------------------
__global__ void fc1_enc_kernel(const float* __restrict__ x,
                               const float* __restrict__ W1,
                               const float* __restrict__ b1,
                               float* __restrict__ cur1) {
    int tid = blockIdx.x * 256 + threadIdx.x;
    if (tid >= BATCH * HID) return;
    int b = tid / HID, h = tid % HID;
    float a = b1[h];
    const float* xr = x + (size_t)b * NIN;
    const float* wr = W1 + (size_t)h * NIN;
#pragma unroll 8
    for (int k = 0; k < NIN; k++) a += xr[k] * wr[k];
    cur1[tid] = a;
}

// ---------------------------------------------------------------------------
// Layer-1 LIF recurrence with constant input current; spikes packed to bits
// via wave32 ballot. Out: [T][B][HID/32] u32.
// ---------------------------------------------------------------------------
__global__ void layer1_spikes_kernel(const float* __restrict__ cur1,
                                     unsigned int* __restrict__ s1p) {
    int tid = blockIdx.x * 256 + threadIdx.x;
    int b = tid / HID, h = tid % HID;
    float cur = cur1[tid];
    float m = 0.0f;
    int w = h >> 5;
    for (int t = 0; t < T_STEPS; t++) {
        float r = (m > THRv) ? THRv : 0.0f;
        m = BETAv * m + cur - r;
        bool s = (m - THRv) > 0.0f;
        unsigned mask = __builtin_amdgcn_ballot_w32(s);
        if ((h & 31) == 0)
            s1p[(size_t)t * (BATCH * KCH) + (size_t)b * KCH + w] = mask;
    }
}

// ---------------------------------------------------------------------------
// Fused layer-2: per-timestep GEMM (split-bf16 WMMA, spikes as A) + LIF
// recurrence held in accumulator registers across all 64 steps.
//   MODE 0 (encoder, N=2048): count spikes -> mean/var -> z (+bf16 split)
//   MODE 1 (decoder, N=40 padded to 48): store spk & mem per step
// ---------------------------------------------------------------------------
template <int BN_TILES, int MODE>
__global__ __launch_bounds__(256) void fused_layer2_kernel(
    const unsigned int* __restrict__ s1p,        // [T][B][KCH] packed spikes
    const unsigned short* __restrict__ Whi,      // [N][HID] bf16 hi
    const unsigned short* __restrict__ Wlo,      // [N][HID] bf16 lo
    const float* __restrict__ bias,              // [Ntrue]
    const float* __restrict__ eps,               // [B][LAT]   (MODE 0)
    float* __restrict__ out_mean,                // [B][LAT]   (MODE 0)
    float* __restrict__ out_var,                 // [B][LAT]   (MODE 0)
    unsigned short* __restrict__ zhi,            // [B][LAT]   (MODE 0)
    unsigned short* __restrict__ zlo,            // [B][LAT]   (MODE 0)
    float* __restrict__ out_spk,                 // [T][B][40] (MODE 1)
    float* __restrict__ out_mem,                 // [T][B][40] (MODE 1)
    int Ntrue) {
    constexpr int BN  = BN_TILES * 16;
    constexpr int NT  = 4 * BN_TILES;        // 16x16 tiles per block
    constexpr int TPW = (NT + 7) / 8;        // tiles per wave

    __shared__ __align__(16) unsigned short lA[BM * LDA];
    __shared__ __align__(16) unsigned short lWhi[BN * LDA];
    __shared__ __align__(16) unsigned short lWlo[BN * LDA];

    const int tid   = threadIdx.x;
    const int lane  = tid & 31;
    const int wave  = tid >> 5;
    const int mBase = blockIdx.x * BM;
    const int nBase = blockIdx.y * BN;

    // Preload W tiles (hi+lo) into LDS once; reused across all 64 timesteps.
    for (int idx = tid; idx < BN * (HID / 8); idx += 256) {
        int r = idx / (HID / 8), c = idx % (HID / 8);
        *(uint4*)&lWhi[r * LDA + c * 8] = *(const uint4*)&Whi[(size_t)(nBase + r) * HID + c * 8];
        *(uint4*)&lWlo[r * LDA + c * 8] = *(const uint4*)&Wlo[(size_t)(nBase + r) * HID + c * 8];
    }

    floatx8 m2[TPW], cnt[TPW];
    float bval[TPW];
#pragma unroll
    for (int s = 0; s < TPW; s++) {
        int ti = wave + s * 8;
        bval[s] = 0.0f;
        if (ti < NT) {
            int tn = ti % BN_TILES;
            int n = nBase + tn * 16 + (lane & 15);
            bval[s] = (n < Ntrue) ? bias[n] : 0.0f;
        }
#pragma unroll
        for (int i = 0; i < 8; i++) { m2[s][i] = 0.0f; cnt[s][i] = 0.0f; }
    }

    for (int t = 0; t < T_STEPS; t++) {
        // Expand this timestep's packed spikes into the bf16 A tile in LDS.
        for (int idx = tid; idx < BM * KCH; idx += 256) {
            int r = idx / KCH, w = idx % KCH;
            unsigned bits = s1p[(size_t)t * (BATCH * KCH) + (size_t)(mBase + r) * KCH + w];
            unsigned int* dst = (unsigned int*)&lA[r * LDA + w * 32];
#pragma unroll
            for (int j = 0; j < 16; j++) {
                unsigned p = (bits & 1u) ? 0x3F80u : 0u;        bits >>= 1;
                p |= (bits & 1u) ? 0x3F800000u : 0u;            bits >>= 1;
                dst[j] = p;
            }
        }
        __syncthreads();

#pragma unroll
        for (int s = 0; s < TPW; s++) {
            int ti = wave + s * 8;
            if (ti < NT) {
                int tm = ti / BN_TILES, tn = ti % BN_TILES;
                floatx8 acc;
#pragma unroll
                for (int i = 0; i < 8; i++) acc[i] = bval[s];
#pragma unroll
                for (int c = 0; c < KCH; c++) {
                    bf16x16 a  = loadA_lds(lA, tm * 16, lane, c);
                    bf16x16 bh = loadB_lds(lWhi, tn * 16, lane, c);
                    bf16x16 bl = loadB_lds(lWlo, tn * 16, lane, c);
                    acc = wmma_bf16(a, bh, acc);
                    acc = wmma_bf16(a, bl, acc);   // residual plane: ~exact f32 product
                }
                // LIF recurrence per owned (m,n) element — stays in registers.
#pragma unroll
                for (int i = 0; i < 8; i++) {
                    float mp = m2[s][i];
                    float r  = (mp > THRv) ? THRv : 0.0f;
                    float nm = BETAv * mp + acc[i] - r;
                    m2[s][i] = nm;
                    float sp = ((nm - THRv) > 0.0f) ? 1.0f : 0.0f;
                    if (MODE == 0) {
                        cnt[s][i] += sp;
                    } else {
                        int n = nBase + tn * 16 + (lane & 15);
                        if (n < Ntrue) {
                            int mg = mBase + tm * 16 + i + ((lane & 16) ? 8 : 0);
                            size_t off = (size_t)t * (BATCH * NIN) + (size_t)mg * NIN + n;
                            out_spk[off] = sp;
                            out_mem[off] = nm;
                        }
                    }
                }
            }
        }
        __syncthreads();
    }

    if (MODE == 0) {
        // Binary spikes: mean = cnt/T, var = mean - mean^2; fuse reparameterize.
#pragma unroll
        for (int s = 0; s < TPW; s++) {
            int ti = wave + s * 8;
            if (ti >= NT) continue;
            int tm = ti / BN_TILES, tn = ti % BN_TILES;
            int n = nBase + tn * 16 + (lane & 15);
#pragma unroll
            for (int i = 0; i < 8; i++) {
                int mg = mBase + tm * 16 + i + ((lane & 16) ? 8 : 0);
                float mean = cnt[s][i] * (1.0f / 64.0f);
                float var  = mean - mean * mean;
                size_t off = (size_t)mg * LAT + n;
                out_mean[off] = mean;
                out_var[off]  = var;
                float std = __expf(0.25f * var);   // sqrt(exp(0.5*var))
                float z   = mean + eps[off] * std;
                unsigned short h = f2bf(z);
                zhi[off] = h;
                zlo[off] = f2bf(z - bf2f(h));
            }
        }
    }
}

// ---------------------------------------------------------------------------
// Decoder fc1: cur1d = z @ dec_W1^T + b1, M=1024 N=320 K=2048, split-bf16 WMMA,
// fragments loaded straight from global (192MB L2 provides reuse).
// ---------------------------------------------------------------------------
__global__ __launch_bounds__(256) void fc1_dec_wmma_kernel(
    const unsigned short* __restrict__ zhi, const unsigned short* __restrict__ zlo,
    const unsigned short* __restrict__ Whi, const unsigned short* __restrict__ Wlo,
    const float* __restrict__ b1, float* __restrict__ cur1d) {
    const int lane = threadIdx.x & 31;
    const int gw   = (blockIdx.x * 256 + threadIdx.x) >> 5;  // global wave id
    const int mt = gw / (HID / 16), nt = gw % (HID / 16);    // 64 x 20 tiles
    const int mBase = mt * 16, nBase = nt * 16;
    const int am = mBase + (lane & 15);
    const int bn = nBase + (lane & 15);

    floatx8 acc;
    float bv = b1[bn];
#pragma unroll
    for (int i = 0; i < 8; i++) acc[i] = bv;

    for (int c = 0; c < LAT / 32; c++) {
        int ka = c * 32 + ((lane & 16) ? 8 : 0);
        int kb = c * 32 + ((lane & 16) ? 8 : 0);
        FragU ah, al, bh, bl;
        ah.q[0] = *(const uint4*)&zhi[(size_t)am * LAT + ka];
        ah.q[1] = *(const uint4*)&zhi[(size_t)am * LAT + ka + 16];
        al.q[0] = *(const uint4*)&zlo[(size_t)am * LAT + ka];
        al.q[1] = *(const uint4*)&zlo[(size_t)am * LAT + ka + 16];
        bh.q[0] = *(const uint4*)&Whi[(size_t)bn * LAT + kb];
        bh.q[1] = *(const uint4*)&Whi[(size_t)bn * LAT + kb + 16];
        bl.q[0] = *(const uint4*)&Wlo[(size_t)bn * LAT + kb];
        bl.q[1] = *(const uint4*)&Wlo[(size_t)bn * LAT + kb + 16];
        acc = wmma_bf16(ah.v, bh.v, acc);
        acc = wmma_bf16(ah.v, bl.v, acc);
        acc = wmma_bf16(al.v, bh.v, acc);   // zl*Wlo term negligible (~2^-16 rel)
    }
#pragma unroll
    for (int i = 0; i < 8; i++) {
        int mg = mBase + i + ((lane & 16) ? 8 : 0);
        cur1d[(size_t)mg * HID + bn] = acc[i];
    }
}

// ---------------------------------------------------------------------------
// Host launcher
// ---------------------------------------------------------------------------
extern "C" void kernel_launch(void* const* d_in, const int* in_sizes, int n_in,
                              void* d_out, int out_size, void* d_ws, size_t ws_size,
                              hipStream_t stream) {
    const float* x    = (const float*)d_in[0];
    const float* eps  = (const float*)d_in[1];
    const float* eW1  = (const float*)d_in[2];
    const float* eb1  = (const float*)d_in[3];
    const float* eW2  = (const float*)d_in[4];
    const float* eb2  = (const float*)d_in[5];
    const float* dW1  = (const float*)d_in[6];
    const float* db1  = (const float*)d_in[7];
    const float* dW2  = (const float*)d_in[8];
    const float* db2  = (const float*)d_in[9];

    char* ws = (char*)d_ws;
    size_t o = 0;
    auto alloc = [&](size_t bytes) -> void* {
        void* p = ws + o;
        o += (bytes + 255) & ~(size_t)255;
        return p;
    };
    float*          cur1e = (float*)alloc((size_t)BATCH * HID * 4);
    unsigned int*   s1e   = (unsigned int*)alloc((size_t)T_STEPS * BATCH * KCH * 4);
    unsigned short* W2hi  = (unsigned short*)alloc((size_t)LAT * HID * 2);
    unsigned short* W2lo  = (unsigned short*)alloc((size_t)LAT * HID * 2);
    unsigned short* zhi   = (unsigned short*)alloc((size_t)BATCH * LAT * 2);
    unsigned short* zlo   = (unsigned short*)alloc((size_t)BATCH * LAT * 2);
    unsigned short* dW1hi = (unsigned short*)alloc((size_t)HID * LAT * 2);
    unsigned short* dW1lo = (unsigned short*)alloc((size_t)HID * LAT * 2);
    float*          cur1d = (float*)alloc((size_t)BATCH * HID * 4);
    unsigned int*   s1d   = (unsigned int*)alloc((size_t)T_STEPS * BATCH * KCH * 4);
    unsigned short* dW2hi = (unsigned short*)alloc((size_t)48 * HID * 2);
    unsigned short* dW2lo = (unsigned short*)alloc((size_t)48 * HID * 2);

    float* out      = (float*)d_out;
    float* out_spk  = out;                                   // [64,1024,40]
    float* out_mem  = out + (size_t)T_STEPS * BATCH * NIN;   // [64,1024,40]
    float* out_mean = out + (size_t)2 * T_STEPS * BATCH * NIN;  // [1024,2048]
    float* out_var  = out_mean + (size_t)BATCH * LAT;           // [1024,2048]

    // 1) weight prep (bf16 hi/lo splits)
    split_bf16_kernel<<<(LAT * HID + 255) / 256, 256, 0, stream>>>(eW2, W2hi, W2lo, LAT * HID);
    split_bf16_kernel<<<(HID * LAT + 255) / 256, 256, 0, stream>>>(dW1, dW1hi, dW1lo, HID * LAT);
    split_pad48_kernel<<<(48 * HID + 255) / 256, 256, 0, stream>>>(dW2, dW2hi, dW2lo);

    // 2) encoder fc1 (constant current) + layer-1 spikes (packed bits)
    fc1_enc_kernel<<<(BATCH * HID) / 256, 256, 0, stream>>>(x, eW1, eb1, cur1e);
    layer1_spikes_kernel<<<(BATCH * HID) / 256, 256, 0, stream>>>(cur1e, s1e);

    // 3) fused encoder layer-2: 64x WMMA GEMM + LIF + mean/var + reparam -> z
    fused_layer2_kernel<4, 0><<<dim3(BATCH / BM, LAT / 64), 256, 0, stream>>>(
        s1e, W2hi, W2lo, eb2, eps, out_mean, out_var, zhi, zlo, nullptr, nullptr, LAT);

    // 4) decoder fc1 (z @ W1^T) via split-bf16 WMMA
    fc1_dec_wmma_kernel<<<(BATCH / 16) * (HID / 16) / 8, 256, 0, stream>>>(
        zhi, zlo, dW1hi, dW1lo, db1, cur1d);

    // 5) decoder layer-1 spikes
    layer1_spikes_kernel<<<(BATCH * HID) / 256, 256, 0, stream>>>(cur1d, s1d);

    // 6) fused decoder layer-2: WMMA GEMM (N padded to 48) + LIF -> spk/mem per t
    fused_layer2_kernel<3, 1><<<dim3(BATCH / BM, 1), 256, 0, stream>>>(
        s1d, dW2hi, dW2lo, db2, nullptr, nullptr, nullptr, nullptr, nullptr,
        out_spk, out_mem, NIN);

    (void)in_sizes; (void)n_in; (void)out_size; (void)ws_size;
}
